// AAGNN_A_82025285419311
// MI455X (gfx1250) — matrix operand order repaired and checked by reading, compiled, not run on hardware
//
#include <hip/hip_runtime.h>

// ---------------------------------------------------------------------------
// AAGNN attention layer, reformulated for MI455X (gfx1250, wave32, WMMA).
//
//   z   = feats @ W^T + b                      (3-term bf16-split WMMA GEMM)
//   e_i = exp(lrelu(zi_i)); d_i = exp(lrelu(zi_i+zj_i))   (per-row scalars)
//   agg = adj @ z          (binary adj exact in bf16; z as hi/lo bf16 split)
//   S_i = e_i*(deg_i-1) + d_i
//   h_i = z_i - (e_i/S_i)*(agg_i - z_i) - (d_i/S_i)*z_i
//   out = relu(h[mask])
//
// eye_matrix (256MB) is never read. adj (256MB) streams exactly once.
// ---------------------------------------------------------------------------

typedef __attribute__((ext_vector_type(16))) __bf16 v16bf;
typedef __attribute__((ext_vector_type(8)))  float  v8f;

#define N_NODES 8192
#define F_IN    512
#define F_OUT   256
#define M_OUT   4096
#define NEG_SLOPE 0.01f

static __device__ __forceinline__ v8f wmma_bf16(v16bf a, v16bf b, v8f c) {
  // D = A(16x32 bf16) x B(32x16 bf16) + C(16x16 f32)
  return __builtin_amdgcn_wmma_f32_16x16x32_bf16(
      /*neg_a=*/false, a, /*neg_b=*/false, b,
      /*c_mod=*/(short)0, c, /*reuse_a=*/false, /*reuse_b=*/false);
}

// Load 8 consecutive f32, split into bf16 hi/lo, place at vector elems [base..base+7].
static __device__ __forceinline__ void cvt8_split(const float* __restrict__ p,
                                                  v16bf& hi, v16bf& lo, int base) {
#pragma unroll
  for (int i = 0; i < 8; i += 4) {
    float4 u = *reinterpret_cast<const float4*>(p + i);
    float f[4] = {u.x, u.y, u.z, u.w};
#pragma unroll
    for (int j = 0; j < 4; j++) {
      __bf16 h = (__bf16)f[j];
      hi[base + i + j] = h;
      lo[base + i + j] = (__bf16)(f[j] - (float)h);
    }
  }
}

// Load 8 consecutive f32 that are exactly representable in bf16 (adj is {0,1});
// also accumulate their sum (row degree).
static __device__ __forceinline__ void cvt8_exact(const float* __restrict__ p,
                                                  v16bf& a, int base, float& rowsum) {
#pragma unroll
  for (int i = 0; i < 8; i += 4) {
    float4 u = *reinterpret_cast<const float4*>(p + i);
    float f[4] = {u.x, u.y, u.z, u.w};
#pragma unroll
    for (int j = 0; j < 4; j++) {
      a[base + i + j] = (__bf16)f[j];
      rowsum += f[j];
    }
  }
}

// ---------------------------------------------------------------------------
// Kernel 1: z = feats @ W^T + b   via 3-term bf16 split (fp32-grade accuracy).
// grid.x = 512 M-tiles; block = 128 (4 waves); each wave owns 4 column tiles.
// ---------------------------------------------------------------------------
__global__ __launch_bounds__(128)
void k_gemm_z(const float* __restrict__ feats, const float* __restrict__ W,
              const float* __restrict__ bias, float* __restrict__ z) {
  const int lane = threadIdx.x & 31;
  const int wave = threadIdx.x >> 5;
  const int m    = lane & 15;    // A row / B column / C column within tile
  const int hh   = lane >> 4;    // lane half (ISA fragment layout)
  const int r0   = blockIdx.x * 16;

  const v8f zero = {0.f, 0.f, 0.f, 0.f, 0.f, 0.f, 0.f, 0.f};
  v8f acc[4] = {zero, zero, zero, zero};

  const float* arow = feats + (size_t)(r0 + m) * F_IN;

  for (int kc = 0; kc < F_IN; kc += 32) {
    v16bf ah, al;
    // A 16x32 layout: lane m holds K = [kc+16h, +16) in elems 0..15.
    cvt8_split(arow + kc + 16 * hh,     ah, al, 0);
    cvt8_split(arow + kc + 16 * hh + 8, ah, al, 8);
#pragma unroll
    for (int t = 0; t < 4; t++) {
      const int colbase = (wave * 4 + t) * 16;
      // B 32x16 layout: lane n holds K = [kc+16h, +16) of column n.
      // B[k][n] = W[colbase+n][k]  -> 16 consecutive floats of a W row.
      const float* wrow = W + (size_t)(colbase + m) * F_IN + kc + 16 * hh;
      v16bf bh, bl;
      cvt8_split(wrow,     bh, bl, 0);
      cvt8_split(wrow + 8, bh, bl, 8);
      acc[t] = wmma_bf16(ah, bh, acc[t]);
      acc[t] = wmma_bf16(ah, bl, acc[t]);
      acc[t] = wmma_bf16(al, bh, acc[t]);
    }
  }

#pragma unroll
  for (int t = 0; t < 4; t++) {
    const int col = (wave * 4 + t) * 16 + m;     // C layout: N = lane%16
    const float bv = bias[col];
#pragma unroll
    for (int v = 0; v < 8; v++) {
      const int grow = r0 + v + 8 * hh;          // C layout: M = v + 8*half
      z[(size_t)grow * F_OUT + col] = acc[t][v] + bv;
    }
  }
}

// ---------------------------------------------------------------------------
// Kernel 2: per-row scalars e,d + hi/lo bf16 split of z stored TRANSPOSED
// (so kernel 3's B-fragments become single contiguous 32B loads per lane).
// ---------------------------------------------------------------------------
__global__ __launch_bounds__(256)
void k_prep(const float* __restrict__ z, const float* __restrict__ a1,
            const float* __restrict__ a2, unsigned short* __restrict__ zT_hi_u,
            unsigned short* __restrict__ zT_lo_u, float* __restrict__ E,
            float* __restrict__ D) {
  const int col = threadIdx.x;
  const int row = blockIdx.x;
  __bf16* zT_hi = reinterpret_cast<__bf16*>(zT_hi_u);
  __bf16* zT_lo = reinterpret_cast<__bf16*>(zT_lo_u);

  const float zv = z[(size_t)row * F_OUT + col];
  const __bf16 hv = (__bf16)zv;
  zT_hi[(size_t)col * N_NODES + row] = hv;
  zT_lo[(size_t)col * N_NODES + row] = (__bf16)(zv - (float)hv);

  __shared__ float s1[256];
  __shared__ float s2[256];
  s1[col] = a1[col] * zv;
  s2[col] = a2[col] * zv;
  __syncthreads();
  for (int st = 128; st > 0; st >>= 1) {
    if (col < st) { s1[col] += s1[col + st]; s2[col] += s2[col + st]; }
    __syncthreads();
  }
  if (col == 0) {
    const float zi = s1[0], zj = s2[0];
    const float p1 = zi > 0.f ? zi : NEG_SLOPE * zi;
    const float sij = zi + zj;
    const float p2 = sij > 0.f ? sij : NEG_SLOPE * sij;
    E[row] = expf(p1);
    D[row] = expf(p2);
  }
}

// ---------------------------------------------------------------------------
// Kernel 3: agg = adj @ z (2 WMMAs/tile: exact bf16 adj x (z_hi + z_lo)),
// degree accumulated for free from the adj fragment, fused finalize:
//   h = z - (e/S)*(agg - z) - (d/S)*z.
// grid.x = 512 M-tiles; block = 128 (4 waves x 4 col-tiles = all 256 cols),
// so adj streams through HBM exactly once (waves in the WGP share the tile
// via cache); zT_hi/zT_lo (8MB) stay resident in the 192MB L2.
// ---------------------------------------------------------------------------
__global__ __launch_bounds__(128)
void k_agg_finalize(const float* __restrict__ adj, const float* __restrict__ z,
                    const unsigned short* __restrict__ zT_hi_u,
                    const unsigned short* __restrict__ zT_lo_u,
                    const float* __restrict__ E, const float* __restrict__ D,
                    float* __restrict__ hbuf) {
  const int lane = threadIdx.x & 31;
  const int wave = threadIdx.x >> 5;
  const int m    = lane & 15;
  const int hh   = lane >> 4;
  const int r0   = blockIdx.x * 16;
  const __bf16* zT_hi = reinterpret_cast<const __bf16*>(zT_hi_u);
  const __bf16* zT_lo = reinterpret_cast<const __bf16*>(zT_lo_u);

  const v8f zero = {0.f, 0.f, 0.f, 0.f, 0.f, 0.f, 0.f, 0.f};
  v8f acc[4] = {zero, zero, zero, zero};
  float degp = 0.f;  // partial row degree held by this lane

  const float* arow = adj + (size_t)(r0 + m) * N_NODES;

  for (int kc = 0; kc < N_NODES; kc += 32) {
    v16bf a;  // adj is {0,1}: exact in bf16, no lo term needed
    cvt8_exact(arow + kc + 16 * hh,     a, 0, degp);
    cvt8_exact(arow + kc + 16 * hh + 8, a, 8, degp);
#pragma unroll
    for (int t = 0; t < 4; t++) {
      const int colbase = (wave * 4 + t) * 16;
      const size_t boff = (size_t)(colbase + m) * N_NODES + kc + 16 * hh;
      const v16bf bh = *reinterpret_cast<const v16bf*>(zT_hi + boff);
      const v16bf bl = *reinterpret_cast<const v16bf*>(zT_lo + boff);
      acc[t] = wmma_bf16(a, bh, acc[t]);
      acc[t] = wmma_bf16(a, bl, acc[t]);
    }
  }

  // lanes m and m+16 each saw half the K range of row r0+m: combine.
  const float degfull = degp + __shfl_xor(degp, 16, 32);

  // Per-output-row attention coefficients (same for every column tile).
  float ec[8], dc[8];
#pragma unroll
  for (int v = 0; v < 8; v++) {
    const int r = v + 8 * hh;
    const int grow = r0 + r;
    const float deg_r = __shfl(degfull, r, 32);  // lane r holds row r's degree
    const float e = E[grow];
    const float d = D[grow];
    const float S = e * (deg_r - 1.0f) + d;      // softmax row sum
    ec[v] = e / S;
    dc[v] = d / S;
  }

#pragma unroll
  for (int t = 0; t < 4; t++) {
    const int col = (wave * 4 + t) * 16 + m;
#pragma unroll
    for (int v = 0; v < 8; v++) {
      const int grow = r0 + v + 8 * hh;
      const float zv = z[(size_t)grow * F_OUT + col];
      const float aggv = acc[t][v];  // sum of z over all neighbors (incl self)
      const float attz = ec[v] * (aggv - zv) + dc[v] * zv;
      hbuf[(size_t)grow * F_OUT + col] = zv - attz;
    }
  }
}

// ---------------------------------------------------------------------------
// Kernel 4: out = relu(h[node_mask])
// ---------------------------------------------------------------------------
__global__ __launch_bounds__(256)
void k_gather(const float* __restrict__ hbuf, const int* __restrict__ mask,
              float* __restrict__ out) {
  const int idx = blockIdx.x * 256 + threadIdx.x;  // grid covers M_OUT*F_OUT
  const int row = idx >> 8;                        // F_OUT == 256
  const int col = idx & 255;
  const int src = mask[row];
  const float v = hbuf[(size_t)src * F_OUT + col];
  out[idx] = v > 0.f ? v : 0.f;
}

// ---------------------------------------------------------------------------
extern "C" void kernel_launch(void* const* d_in, const int* in_sizes, int n_in,
                              void* d_out, int out_size, void* d_ws, size_t ws_size,
                              hipStream_t stream) {
  (void)in_sizes; (void)n_in; (void)out_size; (void)ws_size;
  const float* adj   = (const float*)d_in[0];
  // d_in[1] = eye_matrix: algebraically eliminated, never read (saves 256MB).
  const float* feats = (const float*)d_in[2];
  const int*   mask  = (const int*)d_in[3];
  const float* W     = (const float*)d_in[4];
  const float* bias  = (const float*)d_in[5];
  const float* a1    = (const float*)d_in[6];
  const float* a2    = (const float*)d_in[7];
  float* out = (float*)d_out;

  char* ws = (char*)d_ws;
  float*          z     = (float*)(ws);                                 //  8 MB
  unsigned short* zT_hi = (unsigned short*)(ws + ((size_t)8  << 20));   //  4 MB
  unsigned short* zT_lo = (unsigned short*)(ws + ((size_t)12 << 20));   //  4 MB
  float*          E     = (float*)(ws + ((size_t)16 << 20));            // 32 KB
  float*          D     = (float*)(ws + ((size_t)16 << 20) + 32768);    // 32 KB
  float*          hbuf  = (float*)(ws + ((size_t)17 << 20));            //  8 MB

  k_gemm_z<<<dim3(N_NODES / 16), dim3(128), 0, stream>>>(feats, W, bias, z);
  k_prep<<<dim3(N_NODES), dim3(256), 0, stream>>>(z, a1, a2, zT_hi, zT_lo, E, D);
  k_agg_finalize<<<dim3(N_NODES / 16), dim3(128), 0, stream>>>(adj, z, zT_hi, zT_lo,
                                                               E, D, hbuf);
  k_gather<<<dim3((M_OUT * F_OUT) / 256), dim3(256), 0, stream>>>(hbuf, mask, out);
}
